// PSA_11347303596416
// MI455X (gfx1250) — matrix operand order repaired and checked by reading, not compiled, not run on hardware
//
#include <hip/hip_runtime.h>

// Problem constants (from reference)
#define N_P     16384     // posts
#define MAXLEN  50
#define Dfeat   300
#define Dp      320       // D padded to 10 * 32 (bf16 WMMA K-steps)
#define NFNUM   100
#define NFPAD   112       // NF padded to 7 * 16 N-tiles
#define NCOLS   1344      // 12 (kernel,tap) slices * 112
#define BUSERS  2048

typedef __attribute__((ext_vector_type(16))) __bf16 v16bf;
typedef __attribute__((ext_vector_type(8)))  __bf16 v8bf;
typedef __attribute__((ext_vector_type(8)))  float  v8f;

// Build a v16bf A/B fragment from two contiguous 16-byte chunks
__device__ __forceinline__ v16bf ld16(const __bf16* p0, const __bf16* p1) {
  union { v16bf v; v8bf h[2]; } u;
  u.h[0] = *(const v8bf*)p0;
  u.h[1] = *(const v8bf*)p1;
  return u.v;
}

__device__ __forceinline__ v8f wmma_bf16(v16bf a, v16bf b, v8f c) {
  return __builtin_amdgcn_wmma_f32_16x16x32_bf16(false, a, false, b,
                                                 (short)0, c, false, false);
}

// ---------------------------------------------------------------------------
// Kernel 0: repack conv weights into a WMMA-B-friendly bf16 matrix
// WB[col][d], col = j*112 + nf, j in [0,12): j<3 -> w3 tap j, j<7 -> w4 tap j-3,
// else w5 tap j-7.  Rows d-major so each lane reads two 8-bf16 runs 16 apart.
// ---------------------------------------------------------------------------
__global__ void __launch_bounds__(256)
build_wb_kernel(const float* __restrict__ w3, const float* __restrict__ w4,
                const float* __restrict__ w5, __bf16* __restrict__ WB) {
  int idx = blockIdx.x * 256 + threadIdx.x;       // col*Dp + d
  if (idx >= NCOLS * Dp) return;
  int col = idx / Dp;
  int d   = idx - col * Dp;
  int j   = col / NFPAD;
  int nf  = col - j * NFPAD;
  float v = 0.0f;
  if (nf < NFNUM && d < Dfeat) {
    if (j < 3)      v = w3[(nf * Dfeat + d) * 3 + j];
    else if (j < 7) v = w4[(nf * Dfeat + d) * 4 + (j - 3)];
    else            v = w5[(nf * Dfeat + d) * 5 + (j - 7)];
  }
  WB[idx] = (__bf16)v;
}

// ---------------------------------------------------------------------------
// Kernel 1: fused per-user pipeline. One workgroup (8 waves) per user:
//   segment-mean of embeddings -> bf16 hist tile in LDS [64 x 320]
//   -> WMMA conv (taps as shifted-A GEMMs) -> relu+maxpool -> hist_w proj
//   -> writes rec[b, 64] to d_out.
// ---------------------------------------------------------------------------
__global__ void __launch_bounds__(256)
hist_conv_kernel(const int* __restrict__ tokens, const int* __restrict__ seg,
                 const float* __restrict__ emb, const __bf16* __restrict__ WB,
                 const float* __restrict__ cb3, const float* __restrict__ cb4,
                 const float* __restrict__ cb5, const float* __restrict__ hist_w,
                 const float* __restrict__ hist_b, float* __restrict__ out) {
  __shared__ __align__(16) __bf16 histB[64 * Dp];   // 40960 B
  __shared__ float fbuf[3 * NFNUM + 4];             // pooled features

  const int tid = threadIdx.x;
  const int b   = blockIdx.x;

  // Segment range [start, end) via binary search on sorted segment_ids
  int lo = 0, hi = N_P;
  while (lo < hi) { int m = (lo + hi) >> 1; if (seg[m] < b) lo = m + 1; else hi = m; }
  const int start = lo;
  hi = N_P;
  while (lo < hi) { int m = (lo + hi) >> 1; if (seg[m] < b + 1) lo = m + 1; else hi = m; }
  const int end = lo;
  const float inv = (end > start) ? 1.0f / (float)(end - start) : 0.0f;

  // Zero-fill padded tile (rows 50..63, cols 300..319 stay zero)
  for (int i = tid; i < 64 * Dp; i += 256) histB[i] = (__bf16)0.0f;
  __syncthreads();

  // Segment mean: each thread owns (l,d) pairs, accumulates in a register
  for (int idx = tid; idx < MAXLEN * Dfeat; idx += 256) {
    const int l = idx / Dfeat;
    const int d = idx - l * Dfeat;
    float s = 0.0f;
    for (int p = start; p < end; ++p) {
      const int t = tokens[p * MAXLEN + l];
      s += emb[t * Dfeat + d];
    }
    histB[l * Dp + d] = (__bf16)(s * inv);
  }
  __syncthreads();

  const int wave = tid >> 5;
  const int lane = tid & 31;
  const int half = lane >> 4;   // lane half selects fragment K-offsets
  const int l15  = lane & 15;

  // 21 column groups: 3 conv kernels x 7 N-tiles of 16 output channels
  for (int g = wave; g < 21; g += 8) {
    const int kk    = g / 7;              // 0,1,2 -> K=3,4,5
    const int ntile = g - kk * 7;
    const int Ksz   = 3 + kk;
    const int jbase = (kk == 0) ? 0 : (kk == 1) ? 3 : 7;
    const int nf    = ntile * 16 + l15;   // output channel for this lane column
    const float* bias = (kk == 0) ? cb3 : (kk == 1) ? cb4 : cb5;
    const float bv  = (nf < NFNUM) ? bias[nf] : 0.0f;
    const int tmax  = 50 - Ksz;           // last valid output time index

    v8f acc0 = {}; v8f acc1 = {}; v8f acc2 = {};   // t-tiles 0..15,16..31,32..47
    for (int k = 0; k < Ksz; ++k) {
      // A rows shifted by tap k: row = 16*m + (lane&15) + k  (LDS, per-lane addr)
      const __bf16* arow = &histB[(l15 + k) * Dp + 8 * half];
      // B column for (slice jbase+k, channel nf): two 8-bf16 chunks per lane (k = 8*half.., 16+8*half..)
      const __bf16* bcol = &WB[((jbase + k) * NFPAD + nf) * Dp + 8 * half];
      #pragma unroll
      for (int ds = 0; ds < Dp / 32; ++ds) {
        const int d0 = ds * 32;
        v16bf bb = ld16(bcol + d0, bcol + d0 + 16);
        v16bf a0 = ld16(arow + d0,             arow + d0 + 16);
        v16bf a1 = ld16(arow + 16 * Dp + d0,   arow + 16 * Dp + d0 + 16);
        v16bf a2 = ld16(arow + 32 * Dp + d0,   arow + 32 * Dp + d0 + 16);
        acc0 = wmma_bf16(a0, bb, acc0);
        acc1 = wmma_bf16(a1, bb, acc1);
        acc2 = wmma_bf16(a2, bb, acc2);
      }
    }

    // bias + relu + global max pool (runmax seeded at 0 == implicit relu)
    float runmax = 0.0f;
    #pragma unroll
    for (int q = 0; q < 8; ++q) {
      const int t0 = q + 8 * half;        // C layout: M = q + 8*(lane>=16)
      if (t0      <= tmax) runmax = fmaxf(runmax, acc0[q] + bv);
      if (t0 + 16 <= tmax) runmax = fmaxf(runmax, acc1[q] + bv);
      if (t0 + 32 <= tmax) runmax = fmaxf(runmax, acc2[q] + bv);
    }
    runmax = fmaxf(runmax, __shfl_xor(runmax, 16, 32));  // merge lane halves
    if (half == 0 && nf < NFNUM) fbuf[kk * NFNUM + nf] = runmax;
  }
  __syncthreads();

  // rec = hist_w @ f + hist_b  (64 x 300, tiny)
  if (tid < 64) {
    float o = hist_b[tid];
    const float* wr = hist_w + tid * (3 * NFNUM);
    for (int j = 0; j < 3 * NFNUM; ++j) o += wr[j] * fbuf[j];
    out[b * 64 + tid] = o;
  }
}

// ---------------------------------------------------------------------------
// Kernel 2: fc path in full f32 (0.44 GFLOP, negligible), adds into d_out.
// ---------------------------------------------------------------------------
__global__ void __launch_bounds__(128)
fc_kernel(const float* __restrict__ x, const int* __restrict__ root,
          const float* __restrict__ w1, const float* __restrict__ b1,
          const float* __restrict__ w2, const float* __restrict__ b2,
          float* __restrict__ out) {
  __shared__ float xr[768];
  __shared__ float h[128];
  const int b = blockIdx.x;
  const int r = root[b];
  for (int i = threadIdx.x; i < 768; i += 128) xr[i] = x[r * 768 + i];
  __syncthreads();
  const int j = threadIdx.x;
  float s = b1[j];
  const float* wr = w1 + j * 768;
  for (int i = 0; i < 768; ++i) s += wr[i] * xr[i];
  h[j] = fmaxf(s, 0.0f);
  __syncthreads();
  if (j < 64) {
    float o = b2[j];
    const float* w2r = w2 + j * 128;
    for (int i = 0; i < 128; ++i) o += w2r[i] * h[i];
    out[b * 64 + j] += o;   // rec (kernel 1) + out_fc
  }
}

extern "C" void kernel_launch(void* const* d_in, const int* in_sizes, int n_in,
                              void* d_out, int out_size, void* d_ws, size_t ws_size,
                              hipStream_t stream) {
  const float* x      = (const float*)d_in[0];
  const int*   root   = (const int*)  d_in[1];
  const int*   tokens = (const int*)  d_in[2];
  const int*   seg    = (const int*)  d_in[3];
  const float* emb    = (const float*)d_in[4];
  const float* w3     = (const float*)d_in[5];
  const float* cb3    = (const float*)d_in[6];
  const float* w4     = (const float*)d_in[7];
  const float* cb4    = (const float*)d_in[8];
  const float* w5     = (const float*)d_in[9];
  const float* cb5    = (const float*)d_in[10];
  const float* hist_w = (const float*)d_in[11];
  const float* hist_b = (const float*)d_in[12];
  const float* fc_w1  = (const float*)d_in[13];
  const float* fc_b1  = (const float*)d_in[14];
  const float* fc_w2  = (const float*)d_in[15];
  const float* fc_b2  = (const float*)d_in[16];
  float*  out = (float*)d_out;
  __bf16* WB  = (__bf16*)d_ws;   // 1344*320*2 = 860160 B of scratch

  build_wb_kernel<<<(NCOLS * Dp) / 256, 256, 0, stream>>>(w3, w4, w5, WB);
  hist_conv_kernel<<<BUSERS, 256, 0, stream>>>(tokens, seg, emb, WB,
                                               cb3, cb4, cb5, hist_w, hist_b, out);
  fc_kernel<<<BUSERS, 128, 0, stream>>>(x, root, fc_w1, fc_b1, fc_w2, fc_b2, out);
}
